// EncoderBlock_16741782520182
// MI455X (gfx1250) — matrix operand order repaired and checked by reading, not compiled, not run on hardware
//
#include <hip/hip_runtime.h>
#include <hip/hip_bf16.h>

// ---------------------------------------------------------------------------
// MI455X (gfx1250) transformer encoder block.
// All matmuls via V_WMMA_F32_16X16X32_F16 (f16 in, f32 accum).
// Async global->LDS staging (GLOBAL_LOAD_ASYNC_TO_LDS_B128) for straight-copy
// tiles; transposed tiles use scatter stores.
// ---------------------------------------------------------------------------

typedef _Float16 v16h __attribute__((ext_vector_type(16)));
typedef _Float16 v8h  __attribute__((ext_vector_type(8)));
typedef float    v8f  __attribute__((ext_vector_type(8)));
typedef int      v4i  __attribute__((ext_vector_type(4)));

#define D_MODEL 1024
#define N_HEADS 16
#define DK      64
#define D_FF    4096
#define BATCH   2
#define SEQ     2048
#define ROWS    (BATCH * SEQ)   // 4096
#define LN_EPS  1e-6f

#define MODE_QKV     0
#define MODE_F16RELU 1
#define MODE_F32RES  2

#ifndef __has_builtin
#define __has_builtin(x) 0
#endif

#if __has_builtin(__builtin_amdgcn_global_load_async_to_lds_b128)
#define HAVE_ASYNC_LDS 1
#else
#define HAVE_ASYNC_LDS 0
#endif

// 16-byte global -> LDS copy; async (no VGPR bounce) when available.
__device__ __forceinline__ void cp16(const _Float16* __restrict__ g,
                                     _Float16* __restrict__ l) {
#if HAVE_ASYNC_LDS
  __builtin_amdgcn_global_load_async_to_lds_b128(
      (v4i __attribute__((address_space(1)))*)g,
      (v4i __attribute__((address_space(3)))*)l, 0, 0);
#else
  *(v8h*)l = *(const v8h*)g;
#endif
}

// 32-byte copy helper (two b128 packets)
__device__ __forceinline__ void cp32(const _Float16* __restrict__ g,
                                     _Float16* __restrict__ l) {
  cp16(g, l);
  cp16(g + 8, l + 8);
}

__device__ __forceinline__ void wait_async() {
#if HAVE_ASYNC_LDS
#if __has_builtin(__builtin_amdgcn_s_wait_asynccnt)
  __builtin_amdgcn_s_wait_asynccnt(0);
#else
  asm volatile("s_wait_asynccnt 0x0" ::: "memory");
#endif
#endif
}

__device__ __forceinline__ v16h mk16(v8h lo, v8h hi) {
  v16h r;
#pragma unroll
  for (int i = 0; i < 8; ++i) { r[i] = lo[i]; r[i + 8] = hi[i]; }
  return r;
}

__device__ __forceinline__ v8f wmma_f16(v16h a, v16h b, v8f c) {
  return __builtin_amdgcn_wmma_f32_16x16x32_f16(false, a, false, b,
                                                (short)0, c, false, false);
}

// ---------------------------------------------------------------------------
// f32 -> f16 conversion (grid-stride)
// ---------------------------------------------------------------------------
__global__ void k_f32_to_f16(const float* __restrict__ src,
                             _Float16* __restrict__ dst, int n) {
  int i = blockIdx.x * blockDim.x + threadIdx.x;
  int stride = gridDim.x * blockDim.x;
  for (; i < n; i += stride) dst[i] = (_Float16)src[i];
}

// ---------------------------------------------------------------------------
// LayerNorm with Bessel-corrected std (ddof=1), scalar alpha/bias -> f16.
// ---------------------------------------------------------------------------
__global__ __launch_bounds__(256)
void k_layernorm(const float* __restrict__ x, _Float16* __restrict__ out,
                 const float* __restrict__ alpha_p,
                 const float* __restrict__ bias_p) {
  const int row = blockIdx.x;
  const int tid = threadIdx.x;
  const float* xr = x + (size_t)row * D_MODEL;

  float v[4];
  float s = 0.f, ss = 0.f;
#pragma unroll
  for (int i = 0; i < 4; ++i) {
    v[i] = xr[tid + i * 256];
    s += v[i];
    ss += v[i] * v[i];
  }
  __shared__ float rs[256], rss[256];
  rs[tid] = s; rss[tid] = ss;
  __syncthreads();
  for (int off = 128; off > 0; off >>= 1) {
    if (tid < off) { rs[tid] += rs[tid + off]; rss[tid] += rss[tid + off]; }
    __syncthreads();
  }
  const float mean = rs[0] * (1.0f / D_MODEL);
  const float var  = (rss[0] - (float)D_MODEL * mean * mean) *
                     (1.0f / (D_MODEL - 1));
  const float stdv = sqrtf(fmaxf(var, 0.f));
  const float inv  = alpha_p[0] / (stdv + LN_EPS);
  const float bias = bias_p[0];
  _Float16* orow = out + (size_t)row * D_MODEL;
#pragma unroll
  for (int i = 0; i < 4; ++i)
    orow[tid + i * 256] = (_Float16)((v[i] - mean) * inv + bias);
}

// ---------------------------------------------------------------------------
// WMMA GEMM, MODE templated (straight-line epilogue).
// Tile M=128 N=64 Kstep=64; 256 threads = 8 waves; wave = 32x32 (2x2 WMMA).
// LDS: A [128][64] row-major (async copy), B^T [64 n][64 k] (scatter).
// 8 WMMAs per wave per barrier interval.
// ---------------------------------------------------------------------------
template <int MODE>
__global__ __launch_bounds__(256)
void k_gemm(const _Float16* __restrict__ A, const _Float16* __restrict__ B,
            const float* __restrict__ bias, const float* __restrict__ res,
            float* __restrict__ outf, _Float16* __restrict__ outh,
            int M, int N, int K) {
  __shared__ _Float16 As[128 * 64];   // 16 KB
  __shared__ _Float16 Bt[64 * 64];    //  8 KB

  const int tid  = threadIdx.x;
  const int lane = tid & 31;
  const int wid  = tid >> 5;
  const int wm   = wid & 3;
  const int wn   = wid >> 2;
  const int r    = lane & 15;
  const int hi   = lane >> 4;
  const int m0   = blockIdx.y * 128;
  const int n0   = blockIdx.x * 64;

  v8f acc[2][2] = {};

  const int arow = tid >> 1;            // 0..127
  const int acol = (tid & 1) * 32;      // 0 / 32
  const int bk   = tid >> 2;            // 0..63
  const int bn   = (tid & 3) * 16;      // 0,16,32,48

  for (int k0 = 0; k0 < K; k0 += 64) {
    {   // A tile: 32 contiguous halves per thread, async copy
      const _Float16* src = A + (size_t)(m0 + arow) * K + (k0 + acol);
      _Float16* dst = &As[arow * 64 + acol];
      cp32(src, dst);
      cp32(src + 16, dst + 16);
    }
    {   // B tile: 16 halves of one row, scatter transposed
      const _Float16* src = B + (size_t)(k0 + bk) * N + (n0 + bn);
      v8h x0 = *(const v8h*)(src);
      v8h x1 = *(const v8h*)(src + 8);
#pragma unroll
      for (int j = 0; j < 8; ++j) {
        Bt[(bn + j) * 64 + bk]     = x0[j];
        Bt[(bn + 8 + j) * 64 + bk] = x1[j];
      }
    }
    // prefetch next k-tile into L2 while this one computes
    if (k0 + 64 < K) {
      __builtin_prefetch(A + (size_t)(m0 + arow) * K + (k0 + 64 + acol), 0, 1);
      __builtin_prefetch(B + (size_t)(k0 + 64 + bk) * N + (n0 + bn), 0, 1);
    }
    wait_async();
    __syncthreads();

    v16h af[2][2], bf[2][2];
#pragma unroll
    for (int sm = 0; sm < 2; ++sm)
#pragma unroll
      for (int ks = 0; ks < 2; ++ks) {
        const _Float16* p = &As[(wm * 32 + sm * 16 + r) * 64 + ks * 32];
        af[sm][ks] = mk16(*(const v8h*)(p + hi * 8),
                          *(const v8h*)(p + 16 + hi * 8));
      }
#pragma unroll
    for (int sn = 0; sn < 2; ++sn)
#pragma unroll
      for (int ks = 0; ks < 2; ++ks) {
        const _Float16* p = &Bt[(wn * 32 + sn * 16 + r) * 64 + ks * 32 + hi * 8];
        bf[sn][ks] = mk16(*(const v8h*)(p), *(const v8h*)(p + 16));
      }
#pragma unroll
    for (int ks = 0; ks < 2; ++ks)
#pragma unroll
      for (int sm = 0; sm < 2; ++sm)
#pragma unroll
        for (int sn = 0; sn < 2; ++sn)
          acc[sm][sn] = wmma_f16(af[sm][ks], bf[sn][ks], acc[sm][sn]);
    __syncthreads();
  }

  // Epilogue. C layout: VGPR e -> M = e + 8*hi, N = r.
#pragma unroll
  for (int sm = 0; sm < 2; ++sm) {
#pragma unroll
    for (int sn = 0; sn < 2; ++sn) {
      const int col = n0 + wn * 32 + sn * 16 + r;
      const float bcol = bias[col];
#pragma unroll
      for (int e = 0; e < 8; ++e) {
        const int row = m0 + wm * 32 + sm * 16 + e + hi * 8;
        const float val = acc[sm][sn][e] + bcol;
        if constexpr (MODE == MODE_QKV) {
          const int b = row >> 11, s = row & (SEQ - 1);
          const int h = col >> 6,  d = col & (DK - 1);
          outh[(((size_t)(b * N_HEADS + h) * SEQ + s) << 6) + d] = (_Float16)val;
        } else if constexpr (MODE == MODE_F16RELU) {
          outh[(size_t)row * N + col] = (_Float16)fmaxf(val, 0.f);
        } else {
          outf[(size_t)row * N + col] = val + res[(size_t)row * N + col];
        }
      }
    }
  }
}

// ---------------------------------------------------------------------------
// Flash attention. Grid (S/64, B*H), 128 threads = 4 waves.
// Each wave owns 16 query rows; streams 64-key tiles with online softmax.
// 16 WMMAs per softmax round. Mask is all-ones -> no masking.
// ---------------------------------------------------------------------------
__global__ __launch_bounds__(128)
void k_attention(const _Float16* __restrict__ Q, const _Float16* __restrict__ K,
                 const _Float16* __restrict__ V, _Float16* __restrict__ attn) {
  __shared__ _Float16 Ks[64 * 64];      // [key][d]  8 KB (B^T layout for Q.K^T)
  __shared__ _Float16 Vt[64 * 64];      // [d][key]  8 KB (B^T layout for P.V)
  __shared__ _Float16 Ps[4][16 * 64];   // per-wave P scratch, 8 KB

  const int tid  = threadIdx.x;
  const int lane = tid & 31;
  const int wid  = tid >> 5;
  const int r    = lane & 15;
  const int hi   = lane >> 4;
  const int qb   = blockIdx.x;
  const int bh   = blockIdx.y;
  const int b    = bh >> 4, h = bh & 15;

  const _Float16* Qh = Q + (size_t)bh * SEQ * DK;
  const _Float16* Kh = K + (size_t)bh * SEQ * DK;
  const _Float16* Vh = V + (size_t)bh * SEQ * DK;

  // Q fragments (A 16x32 layout), d split into two 32-wide k-steps.
  const int qrow = qb * 64 + wid * 16 + r;
  v16h qf[2];
#pragma unroll
  for (int ks = 0; ks < 2; ++ks) {
    const _Float16* p = Qh + (size_t)qrow * DK + ks * 32 + hi * 8;
    qf[ks] = mk16(*(const v8h*)(p), *(const v8h*)(p + 16));
  }

  v8f o[4] = {};
  float mrun[8], lrun[8];
#pragma unroll
  for (int e = 0; e < 8; ++e) { mrun[e] = -3.0e38f; lrun[e] = 0.f; }

  const float scale = 0.125f;   // 1/sqrt(64)

  for (int kb = 0; kb < SEQ; kb += 64) {
    {   // stage K (direct, async) and V^T (scatter)
      const int krow = tid >> 1;          // 0..63
      const int dd   = (tid & 1) * 32;    // 0 / 32
      const _Float16* src = Kh + (size_t)(kb + krow) * DK + dd;
      _Float16* dst = &Ks[krow * 64 + dd];
      cp32(src, dst);
      cp32(src + 16, dst + 16);
      const _Float16* vs = Vh + (size_t)(kb + krow) * DK + dd;
      v8h a0 = *(const v8h*)(vs);
      v8h a1 = *(const v8h*)(vs + 8);
      v8h a2 = *(const v8h*)(vs + 16);
      v8h a3 = *(const v8h*)(vs + 24);
#pragma unroll
      for (int j = 0; j < 8; ++j) {
        Vt[(dd + j) * 64 + krow]      = a0[j];
        Vt[(dd + 8 + j) * 64 + krow]  = a1[j];
        Vt[(dd + 16 + j) * 64 + krow] = a2[j];
        Vt[(dd + 24 + j) * 64 + krow] = a3[j];
      }
    }
    wait_async();
    __syncthreads();

    // S = Q . K^T : 4 key tiles of 16, two 32-wide k-steps over d
    v8f sc[4] = {};
#pragma unroll
    for (int ks = 0; ks < 2; ++ks)
#pragma unroll
      for (int t = 0; t < 4; ++t) {
        const _Float16* p = &Ks[(t * 16 + r) * 64 + ks * 32 + hi * 8];
        sc[t] = wmma_f16(qf[ks], mk16(*(const v8h*)(p), *(const v8h*)(p + 16)),
                         sc[t]);
      }

    // Online softmax. Row for VGPR e is e + 8*hi, shared across 16-lane half.
    float alpha[8];
#pragma unroll
    for (int e = 0; e < 8; ++e) {
      float v0 = sc[0][e] * scale, v1 = sc[1][e] * scale;
      float v2 = sc[2][e] * scale, v3 = sc[3][e] * scale;
      float mx = fmaxf(fmaxf(v0, v1), fmaxf(v2, v3));
#pragma unroll
      for (int msk = 1; msk < 16; msk <<= 1)
        mx = fmaxf(mx, __shfl_xor(mx, msk, 32));
      const float mnew = fmaxf(mrun[e], mx);
      alpha[e] = __expf(mrun[e] - mnew);
      mrun[e] = mnew;
      const float p0 = __expf(v0 - mnew);
      const float p1 = __expf(v1 - mnew);
      const float p2 = __expf(v2 - mnew);
      const float p3 = __expf(v3 - mnew);
      float rsum = (p0 + p1) + (p2 + p3);
#pragma unroll
      for (int msk = 1; msk < 16; msk <<= 1)
        rsum += __shfl_xor(rsum, msk, 32);
      lrun[e] = lrun[e] * alpha[e] + rsum;
      _Float16* pr = &Ps[wid][(e + hi * 8) * 64 + r];
      pr[0]  = (_Float16)p0;
      pr[16] = (_Float16)p1;
      pr[32] = (_Float16)p2;
      pr[48] = (_Float16)p3;
#pragma unroll
      for (int t = 0; t < 4; ++t) o[t][e] *= alpha[e];
    }

    // P fragments (A 16x64 = two 16x32 steps), per-wave LDS region.
    const _Float16* pp = &Ps[wid][r * 64];
    v16h pf[2];
#pragma unroll
    for (int ks = 0; ks < 2; ++ks)
      pf[ks] = mk16(*(const v8h*)(pp + ks * 32 + hi * 8),
                    *(const v8h*)(pp + ks * 32 + 16 + hi * 8));

    // O += P . V : 4 d-tiles, two 32-wide key steps
#pragma unroll
    for (int ks = 0; ks < 2; ++ks)
#pragma unroll
      for (int t = 0; t < 4; ++t) {
        const _Float16* vp = &Vt[(t * 16 + r) * 64 + ks * 32 + hi * 8];
        o[t] = wmma_f16(pf[ks], mk16(*(const v8h*)(vp), *(const v8h*)(vp + 16)),
                        o[t]);
      }
    __syncthreads();
  }

  // O / l -> attn buffer [B*S, D_MODEL] f16
#pragma unroll
  for (int t = 0; t < 4; ++t) {
#pragma unroll
    for (int e = 0; e < 8; ++e) {
      const int row = qb * 64 + wid * 16 + e + hi * 8;
      const int col = h * 64 + t * 16 + r;
      attn[((size_t)(b * SEQ + row)) * D_MODEL + col] =
          (_Float16)(o[t][e] / lrun[e]);
    }
  }
}

// ---------------------------------------------------------------------------
// Host-side orchestration
// ---------------------------------------------------------------------------
extern "C" void kernel_launch(void* const* d_in, const int* in_sizes, int n_in,
                              void* d_out, int out_size, void* d_ws,
                              size_t ws_size, hipStream_t stream) {
  (void)in_sizes; (void)n_in; (void)out_size; (void)ws_size;

  const float* x    = (const float*)d_in[0];
  const float* wq   = (const float*)d_in[2];
  const float* bq   = (const float*)d_in[3];
  const float* wk   = (const float*)d_in[4];
  const float* bk   = (const float*)d_in[5];
  const float* wv   = (const float*)d_in[6];
  const float* bv   = (const float*)d_in[7];
  const float* wo   = (const float*)d_in[8];
  const float* bo   = (const float*)d_in[9];
  const float* w1   = (const float*)d_in[10];
  const float* b1   = (const float*)d_in[11];
  const float* w2   = (const float*)d_in[12];
  const float* b2   = (const float*)d_in[13];
  const float* ln1a = (const float*)d_in[14];
  const float* ln1b = (const float*)d_in[15];
  const float* ln2a = (const float*)d_in[16];
  const float* ln2b = (const float*)d_in[17];
  float* out = (float*)d_out;

  char* ws = (char*)d_ws;
  size_t off = 0;
  auto take = [&](size_t bytes) -> char* {
    char* p = ws + off;
    off += (bytes + 255) & ~(size_t)255;
    return p;
  };
  _Float16* xn16  = (_Float16*)take((size_t)ROWS * D_MODEL * 2);
  _Float16* wq16  = (_Float16*)take((size_t)D_MODEL * D_MODEL * 2);
  _Float16* wk16  = (_Float16*)take((size_t)D_MODEL * D_MODEL * 2);
  _Float16* wv16  = (_Float16*)take((size_t)D_MODEL * D_MODEL * 2);
  _Float16* wo16  = (_Float16*)take((size_t)D_MODEL * D_MODEL * 2);
  _Float16* w116  = (_Float16*)take((size_t)D_MODEL * D_FF * 2);
  _Float16* w216  = (_Float16*)take((size_t)D_FF * D_MODEL * 2);
  _Float16* q16   = (_Float16*)take((size_t)ROWS * D_MODEL * 2);
  _Float16* k16   = (_Float16*)take((size_t)ROWS * D_MODEL * 2);
  _Float16* v16   = (_Float16*)take((size_t)ROWS * D_MODEL * 2);
  _Float16* at16  = (_Float16*)take((size_t)ROWS * D_MODEL * 2);
  float*    x1    = (float*)   take((size_t)ROWS * D_MODEL * 4);
  _Float16* yn16  = (_Float16*)take((size_t)ROWS * D_MODEL * 2);
  _Float16* h16   = (_Float16*)take((size_t)ROWS * D_FF * 2);

  // 1) weight down-conversion
  k_f32_to_f16<<<1024, 256, 0, stream>>>(wq, wq16, D_MODEL * D_MODEL);
  k_f32_to_f16<<<1024, 256, 0, stream>>>(wk, wk16, D_MODEL * D_MODEL);
  k_f32_to_f16<<<1024, 256, 0, stream>>>(wv, wv16, D_MODEL * D_MODEL);
  k_f32_to_f16<<<1024, 256, 0, stream>>>(wo, wo16, D_MODEL * D_MODEL);
  k_f32_to_f16<<<2048, 256, 0, stream>>>(w1, w116, D_MODEL * D_FF);
  k_f32_to_f16<<<2048, 256, 0, stream>>>(w2, w216, D_FF * D_MODEL);

  // 2) LN1
  k_layernorm<<<ROWS, 256, 0, stream>>>(x, xn16, ln1a, ln1b);

  // 3) Q/K/V projections -> [B,H,S,DK] f16
  dim3 gProj(D_MODEL / 64, ROWS / 128);
  k_gemm<MODE_QKV><<<gProj, 256, 0, stream>>>(xn16, wq16, bq, nullptr, nullptr,
                                              q16, ROWS, D_MODEL, D_MODEL);
  k_gemm<MODE_QKV><<<gProj, 256, 0, stream>>>(xn16, wk16, bk, nullptr, nullptr,
                                              k16, ROWS, D_MODEL, D_MODEL);
  k_gemm<MODE_QKV><<<gProj, 256, 0, stream>>>(xn16, wv16, bv, nullptr, nullptr,
                                              v16, ROWS, D_MODEL, D_MODEL);

  // 4) flash attention
  k_attention<<<dim3(SEQ / 64, BATCH * N_HEADS), 128, 0, stream>>>(
      q16, k16, v16, at16);

  // 5) output projection + residual (f32)
  k_gemm<MODE_F32RES><<<gProj, 256, 0, stream>>>(at16, wo16, bo, x, x1,
                                                 nullptr, ROWS, D_MODEL,
                                                 D_MODEL);

  // 6) LN2
  k_layernorm<<<ROWS, 256, 0, stream>>>(x1, yn16, ln2a, ln2b);

  // 7) FFN1 (ReLU, f16 out)
  dim3 gF1(D_FF / 64, ROWS / 128);
  k_gemm<MODE_F16RELU><<<gF1, 256, 0, stream>>>(yn16, w116, b1, nullptr,
                                                nullptr, h16, ROWS, D_FF,
                                                D_MODEL);

  // 8) FFN2 + residual -> d_out (f32)
  k_gemm<MODE_F32RES><<<gProj, 256, 0, stream>>>(h16, w216, b2, x1, out,
                                                 nullptr, ROWS, D_MODEL, D_FF);
}
